// MultiHeadAttention_28200755265931
// MI455X (gfx1250) — matrix operand order repaired and checked by reading, not compiled, not run on hardware
//
#include <hip/hip_runtime.h>
#include <stdint.h>

typedef __attribute__((ext_vector_type(16))) __bf16 v16bf;
typedef __attribute__((ext_vector_type(8)))  float  v8f;

#define D_MODEL 1024
#define NHEADS  16
#define DK      64
#define SEQ     2048
#define BATCH   2

// round-to-nearest-even fp32 -> bf16 (raw u16)
static __device__ __forceinline__ unsigned short f2bf(float x) {
    unsigned int u = __float_as_uint(x);
    unsigned int r = u + 0x7FFFu + ((u >> 16) & 1u);
    return (unsigned short)(r >> 16);
}

union FragU { v16bf v; int4 q[2]; };

// ===========================================================================
// GEMM:  Y[M,N] = X[M,K] @ W[N,K]^T + bias[N]      (K = N = 1024, M = 4096)
// Block = 256 thr (8 waves). Block tile 256x128, wave tile 64x64 (16 WMMA).
// Double-buffered LDS: one barrier per K-step; next tile's global loads
// overlap current tile's WMMAs.
// ===========================================================================

static __device__ __forceinline__ void stageA_f32(
    unsigned short* __restrict__ dst, const float4* __restrict__ Ag,
    int mBase, int k0, int tid)
{
    // 256 rows x 32 k  = 2048 float4 chunks, 8 per thread
#pragma unroll
    for (int it = 0; it < 8; ++it) {
        int id  = tid + 256 * it;
        int row = id >> 3;               // 8 float4 per row
        int kq  = (id & 7) * 4;
        float4 f = Ag[((size_t)(mBase + row) * D_MODEL + k0 + kq) >> 2];
        unsigned int w0 = (unsigned int)f2bf(f.x) | ((unsigned int)f2bf(f.y) << 16);
        unsigned int w1 = (unsigned int)f2bf(f.z) | ((unsigned int)f2bf(f.w) << 16);
        *(uint2*)&dst[row * 40 + kq] = make_uint2(w0, w1);
    }
}

static __device__ __forceinline__ void stageA_bf16(
    unsigned short* __restrict__ dst, const uint4* __restrict__ Ag,
    int mBase, int k0, int tid)
{
    // 256 rows x 32 k = 1024 uint4 chunks (8 bf16 each), 4 per thread
#pragma unroll
    for (int it = 0; it < 4; ++it) {
        int id  = tid + 256 * it;
        int row = id >> 2;               // 4 chunks per row
        int kq  = (id & 3) * 8;
        uint4 w = Ag[((size_t)(mBase + row) * D_MODEL + k0 + kq) >> 3];
        *(uint4*)&dst[row * 40 + kq] = w;
    }
}

static __device__ __forceinline__ void stageW_f32(
    unsigned short* __restrict__ dst, const float4* __restrict__ Wg,
    int nBase, int k0, int tid)
{
    // 128 rows x 32 k = 1024 float4 chunks, 4 per thread
#pragma unroll
    for (int it = 0; it < 4; ++it) {
        int id  = tid + 256 * it;
        int row = id >> 3;
        int kq  = (id & 7) * 4;
        float4 f = Wg[((size_t)(nBase + row) * D_MODEL + k0 + kq) >> 2];
        unsigned int w0 = (unsigned int)f2bf(f.x) | ((unsigned int)f2bf(f.y) << 16);
        unsigned int w1 = (unsigned int)f2bf(f.z) | ((unsigned int)f2bf(f.w) << 16);
        *(uint2*)&dst[row * 40 + kq] = make_uint2(w0, w1);
    }
}

template <bool A_BF16, bool PROJ_OUT>
__global__ __launch_bounds__(256) void gemm_bf16_wmma(
    const void* __restrict__ Av, const float* __restrict__ W,
    const float* __restrict__ bias, void* __restrict__ outv)
{
    __shared__ __align__(16) unsigned short Ash[2][256 * 40];
    __shared__ __align__(16) unsigned short Bsh[2][128 * 40];

    const int tid    = threadIdx.x;
    const int lane   = tid & 31;
    const int wave   = tid >> 5;     // 0..7
    const int waveM  = wave >> 1;    // 0..3  (64 rows each)
    const int waveN  = wave & 1;     // 0..1  (64 cols each)
    const int l16    = lane & 15;
    const int halfId = lane >> 4;    // 0/1

    const int nBase = blockIdx.x * 128;
    const int mBase = blockIdx.y * 256;

    v8f acc[4][4];
#pragma unroll
    for (int i = 0; i < 4; ++i)
#pragma unroll
        for (int j = 0; j < 4; ++j) acc[i][j] = v8f{};

    // prologue: stage k-step 0 into buffer 0
    if (A_BF16) stageA_bf16(Ash[0], (const uint4*)Av, mBase, 0, tid);
    else        stageA_f32 (Ash[0], (const float4*)Av, mBase, 0, tid);
    stageW_f32(Bsh[0], (const float4*)W, nBase, 0, tid);

    const int NK = D_MODEL / 32;     // 32 k-steps
    for (int ki = 0; ki < NK; ++ki) {
        __syncthreads();
        if (ki + 1 < NK) {
            int b = (ki + 1) & 1;
            int k0 = (ki + 1) * 32;
            if (A_BF16) stageA_bf16(Ash[b], (const uint4*)Av, mBase, k0, tid);
            else        stageA_f32 (Ash[b], (const float4*)Av, mBase, k0, tid);
            stageW_f32(Bsh[b], (const float4*)W, nBase, k0, tid);
        }
        const unsigned short* As = Ash[ki & 1];
        const unsigned short* Bs = Bsh[ki & 1];

        // A fragments: lane = M, K groups {half*8..+7, 16+half*8..+7}
        v16bf af[4];
#pragma unroll
        for (int ms = 0; ms < 4; ++ms) {
            int row  = waveM * 64 + ms * 16 + l16;
            int base = row * 40 + halfId * 8;
            FragU u;
            u.q[0] = *(const int4*)&As[base];
            u.q[1] = *(const int4*)&As[base + 16];
            af[ms] = u.v;
        }
        // B fragments one at a time (register pressure); lane = N, K groups {half*8..+7, 16+half*8..+7}
#pragma unroll
        for (int nt = 0; nt < 4; ++nt) {
            int row  = waveN * 64 + nt * 16 + l16;
            int base = row * 40 + halfId * 8;
            FragU u;
            u.q[0] = *(const int4*)&Bs[base];
            u.q[1] = *(const int4*)&Bs[base + 16];
#pragma unroll
            for (int ms = 0; ms < 4; ++ms)
                acc[ms][nt] = __builtin_amdgcn_wmma_f32_16x16x32_bf16(
                    false, af[ms], false, u.v, (short)0, acc[ms][nt],
                    false, false);
        }
    }

    // epilogue: C layout lane = N, row = r + half*8 per VGPR r
#pragma unroll
    for (int ms = 0; ms < 4; ++ms) {
#pragma unroll
        for (int nt = 0; nt < 4; ++nt) {
            int   n  = nBase + waveN * 64 + nt * 16 + l16;
            float bv = bias[n];
#pragma unroll
            for (int r = 0; r < 8; ++r) {
                int   m   = mBase + waveM * 64 + ms * 16 + r + halfId * 8;
                float val = acc[ms][nt][r] + bv;
                if (PROJ_OUT) {
                    unsigned short* O = (unsigned short*)outv;
                    int b = m >> 11, s = m & 2047;
                    int h = n >> 6,  dk = n & 63;
                    O[((((size_t)b * NHEADS + h) * SEQ + s) << 6) + dk] = f2bf(val);
                } else {
                    float* O = (float*)outv;
                    O[(size_t)m * D_MODEL + n] = val;
                }
            }
        }
    }
}

// ===========================================================================
// Flash attention over head-major bf16 Q/K/V [bh, s, 64].
// Block = 128 thr (4 waves); block = 64 query rows, wave = 16 rows.
// Double-buffered 64-key K/V tiles; V stored transposed for P*V.
// ===========================================================================

static __device__ __forceinline__ void stageK_tile(
    unsigned short* __restrict__ dst, const uint4* __restrict__ Kg, int tid)
{
#pragma unroll
    for (int it = 0; it < 4; ++it) {
        int id  = tid + 128 * it;        // 0..511
        int row = id >> 3;               // 8 uint4 per 64-elem row
        int dq  = (id & 7) * 8;
        *(uint4*)&dst[row * 72 + dq] = Kg[(row * DK + dq) >> 3];
    }
}

static __device__ __forceinline__ void stageV_tile_t(
    unsigned short* __restrict__ dst, const uint4* __restrict__ Vg, int tid)
{
#pragma unroll
    for (int it = 0; it < 4; ++it) {
        int id  = tid + 128 * it;
        int row = id >> 3;
        int dq  = (id & 7) * 8;
        uint4 v = Vg[(row * DK + dq) >> 3];
        unsigned int wv[4] = {v.x, v.y, v.z, v.w};
#pragma unroll
        for (int j = 0; j < 4; ++j) {
            dst[(dq + 2 * j + 0) * 72 + row] = (unsigned short)(wv[j] & 0xFFFFu);
            dst[(dq + 2 * j + 1) * 72 + row] = (unsigned short)(wv[j] >> 16);
        }
    }
}

__global__ __launch_bounds__(128) void flash_attn_wmma(
    const unsigned short* __restrict__ Qh,
    const unsigned short* __restrict__ Kh,
    const unsigned short* __restrict__ Vh,
    unsigned short* __restrict__ AT)
{
    __shared__ __align__(16) unsigned short Ksh[2][64 * 72];
    __shared__ __align__(16) unsigned short Vts[2][64 * 72];
    __shared__ __align__(16) unsigned short Psh[4][16 * 72];

    const int tid    = threadIdx.x;
    const int lane   = tid & 31;
    const int wave   = tid >> 5;   // 0..3
    const int l16    = lane & 15;
    const int halfId = lane >> 4;

    const int qt = blockIdx.x;     // query tile (64 rows)
    const int bh = blockIdx.y;     // batch*head

    const unsigned short* Qp = Qh + ((size_t)bh * SEQ + (size_t)qt * 64) * DK;
    const unsigned short* Kp = Kh + (size_t)bh * SEQ * DK;
    const unsigned short* Vp = Vh + (size_t)bh * SEQ * DK;

    // ---- stage Q tile through Ksh[0], build per-wave A fragments ----
    stageK_tile(Ksh[0], (const uint4*)Qp, tid);
    __syncthreads();
    v16bf qf[2];
#pragma unroll
    for (int ks = 0; ks < 2; ++ks) {
        int row  = wave * 16 + l16;
        int base = row * 72 + ks * 32 + halfId * 8;
        FragU u;
        u.q[0] = *(const int4*)&Ksh[0][base];
        u.q[1] = *(const int4*)&Ksh[0][base + 16];
        qf[ks] = u.v;
    }
    __syncthreads();   // all waves done reading Q before K tile 0 overwrites

    // prologue: stage key tile 0 into buffer 0
    stageK_tile  (Ksh[0], (const uint4*)Kp, tid);
    stageV_tile_t(Vts[0], (const uint4*)Vp, tid);

    v8f o_acc[4];
#pragma unroll
    for (int i = 0; i < 4; ++i) o_acc[i] = v8f{};
    float m_row[8], l_row[8];
#pragma unroll
    for (int r = 0; r < 8; ++r) { m_row[r] = -1e30f; l_row[r] = 0.f; }

    const int NT = SEQ / 64;       // 32 key tiles
    for (int kt = 0; kt < NT; ++kt) {
        __syncthreads();
        if (kt + 1 < NT) {
            int b = (kt + 1) & 1;
            stageK_tile  (Ksh[b], (const uint4*)(Kp + (size_t)(kt + 1) * 64 * DK), tid);
            stageV_tile_t(Vts[b], (const uint4*)(Vp + (size_t)(kt + 1) * 64 * DK), tid);
        }
        const unsigned short* Ks = Ksh[kt & 1];
        const unsigned short* Vs = Vts[kt & 1];

        // ---- S = Q K^T  (4 key sub-tiles of 16, K-dim = d = 2x32) ----
        v8f st[4];
#pragma unroll
        for (int nt = 0; nt < 4; ++nt) {
            v8f c = v8f{};
#pragma unroll
            for (int ks = 0; ks < 2; ++ks) {
                int key  = nt * 16 + l16;
                int base = key * 72 + ks * 32 + halfId * 8;
                FragU u;
                u.q[0] = *(const int4*)&Ks[base];
                u.q[1] = *(const int4*)&Ks[base + 16];
                c = __builtin_amdgcn_wmma_f32_16x16x32_bf16(
                        false, qf[ks], false, u.v, (short)0, c, false, false);
            }
            st[nt] = c * 0.125f;   // 1/sqrt(64)
        }

        // ---- online softmax: each row lives in the 16 lanes of a half ----
#pragma unroll
        for (int r = 0; r < 8; ++r) {
            float mx = fmaxf(fmaxf(st[0][r], st[1][r]), fmaxf(st[2][r], st[3][r]));
#pragma unroll
            for (int off = 1; off < 16; off <<= 1)
                mx = fmaxf(mx, __shfl_xor(mx, off, 32));
            float m_new = fmaxf(m_row[r], mx);
            float alpha = __expf(m_row[r] - m_new);
            float rs = 0.f;
#pragma unroll
            for (int nt = 0; nt < 4; ++nt) {
                float p = __expf(st[nt][r] - m_new);
                st[nt][r] = p;
                rs += p;
            }
#pragma unroll
            for (int off = 1; off < 16; off <<= 1)
                rs += __shfl_xor(rs, off, 32);
            l_row[r] = l_row[r] * alpha + rs;
            m_row[r] = m_new;
#pragma unroll
            for (int dt = 0; dt < 4; ++dt) o_acc[dt][r] *= alpha;
        }

        // ---- P: C-layout -> A-layout via per-wave LDS round trip ----
        unsigned short* Pw = &Psh[wave][0];
#pragma unroll
        for (int nt = 0; nt < 4; ++nt)
#pragma unroll
            for (int r = 0; r < 8; ++r) {
                int row = r + halfId * 8;
                Pw[row * 72 + nt * 16 + l16] = f2bf(st[nt][r]);
            }
        v16bf pf[2];
#pragma unroll
        for (int ks = 0; ks < 2; ++ks) {
            int base = l16 * 72 + ks * 32 + halfId * 8;
            FragU u;
            u.q[0] = *(const int4*)&Pw[base];
            u.q[1] = *(const int4*)&Pw[base + 16];
            pf[ks] = u.v;
        }

        // ---- O += P * V  (4 d sub-tiles, K-dim = keys = 2x32) ----
#pragma unroll
        for (int dt = 0; dt < 4; ++dt) {
#pragma unroll
            for (int ks = 0; ks < 2; ++ks) {
                int d    = dt * 16 + l16;
                int base = d * 72 + ks * 32 + halfId * 8;
                FragU u;
                u.q[0] = *(const int4*)&Vs[base];
                u.q[1] = *(const int4*)&Vs[base + 16];
                o_acc[dt] = __builtin_amdgcn_wmma_f32_16x16x32_bf16(
                        false, pf[ks], false, u.v, (short)0, o_acc[dt],
                        false, false);
            }
        }
    }

    // ---- epilogue: divide by row sum, write bf16 AT[b, s, h*64+d] ----
    const int b = bh >> 4, h = bh & 15;
#pragma unroll
    for (int r = 0; r < 8; ++r) {
        float inv = 1.0f / l_row[r];
        int   q   = qt * 64 + wave * 16 + r + halfId * 8;
#pragma unroll
        for (int dt = 0; dt < 4; ++dt) {
            int col = h * 64 + dt * 16 + l16;
            AT[((size_t)b * SEQ + q) * D_MODEL + col] = f2bf(o_acc[dt][r] * inv);
        }
    }
}

// ===========================================================================
extern "C" void kernel_launch(void* const* d_in, const int* in_sizes, int n_in,
                              void* d_out, int out_size, void* d_ws, size_t ws_size,
                              hipStream_t stream)
{
    const float* q   = (const float*)d_in[0];
    const float* k   = (const float*)d_in[1];
    const float* v   = (const float*)d_in[2];
    const float* w_q = (const float*)d_in[3];
    const float* b_q = (const float*)d_in[4];
    const float* w_k = (const float*)d_in[5];
    const float* b_k = (const float*)d_in[6];
    const float* w_v = (const float*)d_in[7];
    const float* b_v = (const float*)d_in[8];
    const float* w_o = (const float*)d_in[9];
    const float* b_o = (const float*)d_in[10];

    char* ws = (char*)d_ws;
    unsigned short* qh = (unsigned short*)(ws);                            // 8 MB
    unsigned short* kh = (unsigned short*)(ws + (size_t)8  * 1024 * 1024); // 8 MB
    unsigned short* vh = (unsigned short*)(ws + (size_t)16 * 1024 * 1024); // 8 MB
    unsigned short* at = (unsigned short*)(ws + (size_t)24 * 1024 * 1024); // 8 MB

    dim3 gg(D_MODEL / 128, (BATCH * SEQ) / 256);   // (8, 16)
    dim3 gb(256);

    gemm_bf16_wmma<false, true><<<gg, gb, 0, stream>>>(q, w_q, b_q, qh);
    gemm_bf16_wmma<false, true><<<gg, gb, 0, stream>>>(k, w_k, b_k, kh);
    gemm_bf16_wmma<false, true><<<gg, gb, 0, stream>>>(v, w_v, b_v, vh);

    flash_attn_wmma<<<dim3(SEQ / 64, BATCH * NHEADS), 128, 0, stream>>>(qh, kh, vh, at);

    gemm_bf16_wmma<true, false><<<gg, gb, 0, stream>>>(at, w_o, b_o, d_out);
}
